// DTVNet_60610578482023
// MI455X (gfx1250) — matrix operand order repaired and checked by reading, not compiled, not run on hardware
//
#include <hip/hip_runtime.h>
#include <hip/hip_bf16.h>
#include <stdint.h>

// ---------------- problem constants ----------------
#define DD 64
#define HH 256
#define WW 256
#define NN (DD*HH*WW)          // 4,194,304 voxels per volume

typedef _Float16 v16h __attribute__((ext_vector_type(16)));
typedef float    v8f  __attribute__((ext_vector_type(8)));
typedef unsigned int u32x4 __attribute__((ext_vector_type(4)));
typedef int          i32x8 __attribute__((ext_vector_type(8)));
typedef int          i32x4 __attribute__((ext_vector_type(4)));

#ifndef __has_builtin
#define __has_builtin(x) 0
#endif
#if __has_builtin(__builtin_amdgcn_tensor_load_to_lds)
#define HAVE_TDM 1
#else
#define HAVE_TDM 0
#endif
#if __has_builtin(__builtin_amdgcn_s_wait_tensorcnt)
#define WAIT_TENSOR0() __builtin_amdgcn_s_wait_tensorcnt(0)
#else
#define WAIT_TENSOR0() asm volatile("s_wait_tensorcnt 0x0" ::: "memory")
#endif

#if HAVE_TDM
// Issue one TDM tensor_load_to_lds for a [tz][ty][tx] tile (X contiguous)
// of a [DD][HH][WW]-shaped tensor. dsz_code: 1 = 2-byte elems, 2 = 4-byte.
// dim3 (channels) optional: nzz tiles of stride d2stride elements.
__device__ __forceinline__ void tdm_tile_load(unsigned long long ga,
                                              unsigned int ldso, int dsz_code,
                                              int tx, int ty, int tz,
                                              int nzz, long d2stride) {
  u32x4 g0 = { 1u, ldso, (unsigned int)ga,
               (unsigned int)(ga >> 32) | (2u << 30) };          // type=2
  i32x8 g1 = { dsz_code << 16,             // workgroup_mask=0 | data_size
               (int)(WW << 16),            // tensor_dim0 lo16
               (int)(HH << 16),            // dim0 hi16(0) | tensor_dim1 lo16
               (int)(tx << 16),            // dim1 hi16(0) | tile_dim0
               (int)(ty | (tz << 16)),     // tile_dim1 | tile_dim2
               WW,                         // tensor_dim0_stride lo32
               0,                          // stride hi16 | dim1_stride lo16
               (int)((HH * WW) >> 16) };   // tensor_dim1_stride >> 16
  i32x4 g2 = { DD,                         // tensor_dim2
               nzz,                        // tensor_dim3 (0 = unused)
               (int)(d2stride & 0xffffffffL),         // dim2_stride lo32
               (int)(((d2stride >> 32) & 0xffff) | ((long)nzz << 16)) }; // hi16 | tile_dim3
  i32x4 g3 = { 0, 0, 0, 0 };
#if defined(__clang_major__) && (__clang_major__ >= 23)
  i32x8 g4 = { 0, 0, 0, 0, 0, 0, 0, 0 };
  __builtin_amdgcn_tensor_load_to_lds(g0, g1, g2, g3, g4, 0);
#else
  __builtin_amdgcn_tensor_load_to_lds(g0, g1, g2, g3, 0);
#endif
}
#endif

// ---------------- helper kernels ----------------

__global__ void zero_kernel(float* __restrict__ buf, long n) {
  long i = (long)blockIdx.x * blockDim.x + threadIdx.x;
  if (i < n) buf[i] = 0.0f;
}

// z[d,y,x] = t[d,y,x] + (sino[y,x] - sum_d t[d,y,x]) / D   (SIRT step)
__global__ void proj_z_kernel(const float* __restrict__ t,
                              const float* __restrict__ sino,
                              float* __restrict__ z) {
  int i = blockIdx.x * blockDim.x + threadIdx.x;   // over H*W
  if (i >= HH * WW) return;
  float sum = 0.0f;
  for (int d = 0; d < DD; ++d) sum += t[(size_t)d * HH * WW + i];
  float r = (sino[i] - sum) * (1.0f / (float)DD);
  for (int d = 0; d < DD; ++d)
    z[(size_t)d * HH * WW + i] = t[(size_t)d * HH * WW + i] + r;
}

// t_new = fdiff_t(p,x) + fdiff_t(q,y) + fdiff_t(s,d) + z_
// fdiff_t(y)[j] = (j>0 ? y[j-1] : 0) - (j<n-1 ? y[j] : 0)
__global__ void assemble_kernel(const float* __restrict__ p,
                                const float* __restrict__ q,
                                const float* __restrict__ s,
                                const float* __restrict__ zb,
                                float* __restrict__ tout) {
  long i = (long)blockIdx.x * blockDim.x + threadIdx.x;
  if (i >= NN) return;
  int x = (int)(i % WW);
  int y = (int)((i / WW) % HH);
  int d = (int)(i / ((long)HH * WW));
  float r = zb[i];
  r += (x > 0 ? p[i - 1]           : 0.0f) - (x < WW - 1 ? p[i] : 0.0f);
  r += (y > 0 ? q[i - WW]          : 0.0f) - (y < HH - 1 ? q[i] : 0.0f);
  r += (d > 0 ? s[i - (long)HH*WW] : 0.0f) - (d < DD - 1 ? s[i] : 0.0f);
  tout[i] = r;
}

// ---------------- conv1 (1 -> 8 ch) via WMMA ----------------
// One v_wmma_f32_16x16x32_f16 per 16-voxel x-row computes all 8 output
// channels over all 27 taps (K=32 >= 27). fdiff applied on the fly.
// MODE: 0 = identity, 1 = d/dx, 2 = d/dy, 3 = d/dz
template <int MODE>
__global__ void __launch_bounds__(256)
conv1_wmma_kernel(const float* __restrict__ zin, const float* __restrict__ w1,
                  const float* __restrict__ b1, _Float16* __restrict__ hout) {
  constexpr int TX = 16, TY = 16, TD = 4;   // output tile
  constexpr int LX = 19, LY = 19, LZ = 7;   // input tile with halo (-1 .. +2)
  __shared__ float zt[LZ * LY * LX];

  const int x0 = blockIdx.x * TX, y0 = blockIdx.y * TY, d0 = blockIdx.z * TD;
  const int gx0 = x0 - 1, gy0 = y0 - 1, gd0 = d0 - 1;

  bool tdm_done = false;
#if HAVE_TDM
  // Interior tile: one Tensor-Data-Mover descriptor loads the whole
  // 7x19x19 f32 sub-volume into LDS (TENSORcnt-tracked).
  const bool interior = (gd0 >= 0) && (gd0 + LZ <= DD) &&
                        (gy0 >= 0) && (gy0 + LY <= HH) &&
                        (gx0 >= 0) && (gx0 + LX <= WW);
  if (interior) {
    if (threadIdx.x < 32) {   // one wave issues the TDM op
      unsigned long long ga = (unsigned long long)(size_t)
          (zin + ((size_t)gd0 * HH + gy0) * WW + gx0);
      unsigned int ldso = (unsigned int)(size_t)(&zt[0]); // low 32b = LDS offset
      tdm_tile_load(ga, ldso, /*dsz=4B*/2, LX, LY, LZ, /*nzz*/0, 0);
      WAIT_TENSOR0();
    }
    tdm_done = true;
  }
#endif
  if (!tdm_done) {  // boundary tiles: cooperative zero-padded load
    for (int i = threadIdx.x; i < LZ * LY * LX; i += 256) {
      int lx = i % LX; int t1 = i / LX; int ly = t1 % LY; int lz = t1 / LY;
      int gx = gx0 + lx, gy = gy0 + ly, gd = gd0 + lz;
      float val0 = 0.0f;
      if ((unsigned)gx < WW && (unsigned)gy < HH && (unsigned)gd < DD)
        val0 = zin[((size_t)gd * HH + gy) * WW + gx];
      zt[i] = val0;
    }
  }
  __syncthreads();

  const int lane  = threadIdx.x & 31;
  const int wavei = threadIdx.x >> 5;
  const int mrow  = lane & 15;   // A: channel row / B: voxel column (x)
  const int hi    = lane >> 4;

  // A fragment: weights [8 x 27] padded to 16x32 (16-bit A layout).
  // Unconditional loads from clamped addresses + select (no divergence).
  v16h afrag;
#pragma unroll
  for (int v = 0; v < 8; ++v) {
#pragma unroll
    for (int s2 = 0; s2 < 2; ++s2) {
      int k = (v < 4 ? 0 : 16) + 8 * hi + ((v & 3) << 1) + s2;
      int kc = k < 27 ? k : 26;
      float wv = w1[(mrow & 7) * 27 + kc];
      afrag[2 * v + s2] = (_Float16)((mrow < 8 && k < 27) ? wv : 0.0f);
    }
  }
  float bias[8];
#pragma unroll
  for (int r = 0; r < 8; ++r) bias[r] = b1[r];

  // 64 x-row groups per tile, 8 per wave
  for (int gi = 0; gi < 8; ++gi) {
    int g = wavei * 8 + gi;
    int yl = g & 15, dl = g >> 4;
    v16h bfrag;
#pragma unroll
    for (int v = 0; v < 8; ++v) {
#pragma unroll
      for (int s2 = 0; s2 < 2; ++s2) {
        int k = (v < 4 ? 0 : 16) + 8 * hi + ((v & 3) << 1) + s2;      // dense 16-bit B layout
        // tap decomposition; for the K-pad slots (27..31) the indices are
        // still inside the haloed LDS tile, so loads are unconditional.
        int kz = k / 9, kr = k - kz * 9, ky = kr / 3, kx = kr - ky * 3;
        int idx = ((dl + kz) * LY + (yl + ky)) * LX + (mrow + kx);
        float val;
        if (MODE == 0) {
          val = zt[idx];
        } else if (MODE == 1) {             // fdiff along x
          int gxt = x0 + mrow + kx - 1;
          float dv = zt[idx + 1] - zt[idx];
          val = ((unsigned)gxt < (WW - 1)) ? dv : 0.0f;
        } else if (MODE == 2) {             // fdiff along y
          int gyt = y0 + yl + ky - 1;
          float dv = zt[idx + LX] - zt[idx];
          val = ((unsigned)gyt < (HH - 1)) ? dv : 0.0f;
        } else {                            // fdiff along d
          int gdt = d0 + dl + kz - 1;
          float dv = zt[idx + LY * LX] - zt[idx];
          val = ((unsigned)gdt < (DD - 1)) ? dv : 0.0f;
        }
        bfrag[2 * v + s2] = (_Float16)(k < 27 ? val : 0.0f);
      }
    }
    v8f acc = {0.f, 0.f, 0.f, 0.f, 0.f, 0.f, 0.f, 0.f};
    acc = __builtin_amdgcn_wmma_f32_16x16x32_f16(
        false, afrag, false, bfrag, (short)0, acc, false, false);
    if (hi == 0) {  // rows M=0..7 (channels) live in lanes 0..15, VGPRs 0..7
      size_t vidx = ((size_t)(d0 + dl) * HH + (y0 + yl)) * WW + (x0 + mrow);
#pragma unroll
      for (int r = 0; r < 8; ++r) {
        float o = acc[r] + bias[r];
        o = o > 0.0f ? o : 0.0f;            // ReLU
        hout[(size_t)r * NN + vidx] = (_Float16)o;
      }
    }
  }
}

// ---------------- conv2 (8 -> 1 ch) via WMMA + fused recurrence ----------------
// K = 8*27 = 216 -> 7 k-chunks of v_wmma_f32_16x16x32_f16 accumulating in C.
// Epilogue: upd = (1+eta)*upd - eta*out   (p/q/s update), or
//           upd = (1+eta)*t   - eta*out   (z_ = t + nt*(t - znew)) when isZ.
__global__ void __launch_bounds__(256)
conv2_wmma_kernel(const _Float16* __restrict__ hin, const float* __restrict__ w2,
                  const float* __restrict__ b2, float* __restrict__ upd,
                  const float* __restrict__ tcur, const float* __restrict__ eta,
                  int isZ) {
  constexpr int TX = 16, TY = 16, TD = 4;
  constexpr int LX = 18, LY = 18, LZ = 6;  // halo 1 each side
  __shared__ _Float16 ht[8 * LZ * LY * LX];

  const int x0 = blockIdx.x * TX, y0 = blockIdx.y * TY, d0 = blockIdx.z * TD;
  const int gx0 = x0 - 1, gy0 = y0 - 1, gd0 = d0 - 1;

  bool tdm_done = false;
#if HAVE_TDM
  // Interior tile: one 4D TDM descriptor: 18x18x6 x 8 channels of f16,
  // channel stride = NN elements.
  const bool interior = (gd0 >= 0) && (gd0 + LZ <= DD) &&
                        (gy0 >= 0) && (gy0 + LY <= HH) &&
                        (gx0 >= 0) && (gx0 + LX <= WW);
  if (interior) {
    if (threadIdx.x < 32) {
      unsigned long long ga = (unsigned long long)(size_t)
          (hin + ((size_t)gd0 * HH + gy0) * WW + gx0);
      unsigned int ldso = (unsigned int)(size_t)(&ht[0]);
      tdm_tile_load(ga, ldso, /*dsz=2B*/1, LX, LY, LZ, /*nzz*/8, (long)NN);
      WAIT_TENSOR0();
    }
    tdm_done = true;
  }
#endif
  if (!tdm_done) {
    for (int i = threadIdx.x; i < 8 * LZ * LY * LX; i += 256) {
      int lx = i % LX; int t1 = i / LX; int ly = t1 % LY; t1 /= LY;
      int lz = t1 % LZ; int c = t1 / LZ;
      int gx = gx0 + lx, gy = gy0 + ly, gd = gd0 + lz;
      _Float16 val0 = (_Float16)0.0f;
      if ((unsigned)gx < WW && (unsigned)gy < HH && (unsigned)gd < DD)
        val0 = hin[(size_t)c * NN + ((size_t)gd * HH + gy) * WW + gx];
      ht[i] = val0;
    }
  }
  __syncthreads();

  const int lane  = threadIdx.x & 31;
  const int wavei = threadIdx.x >> 5;
  const int mrow  = lane & 15;
  const int hi    = lane >> 4;

  // A fragments: w2 flat [216] in row 0, rows 1..15 zero; 7 chunks of K=32
  v16h aw[7];
#pragma unroll
  for (int j = 0; j < 7; ++j) {
#pragma unroll
    for (int v = 0; v < 8; ++v) {
#pragma unroll
      for (int s2 = 0; s2 < 2; ++s2) {
        int k = (v < 4 ? 0 : 16) + 8 * hi + ((v & 3) << 1) + s2;
        int kk = 32 * j + k;
        float wv = w2[kk < 216 ? kk : 215];
        aw[j][2 * v + s2] = (_Float16)((mrow == 0 && kk < 216) ? wv : 0.0f);
      }
    }
  }
  float bb = b2[0];
  float e  = eta[0];

  for (int gi = 0; gi < 8; ++gi) {
    int g = wavei * 8 + gi;
    int yl = g & 15, dl = g >> 4;
    v8f acc = {0.f, 0.f, 0.f, 0.f, 0.f, 0.f, 0.f, 0.f};
#pragma unroll
    for (int j = 0; j < 7; ++j) {
      v16h bfrag;
#pragma unroll
      for (int v = 0; v < 8; ++v) {
#pragma unroll
        for (int s2 = 0; s2 < 2; ++s2) {
          int kk = 32 * j + (v < 4 ? 0 : 16) + 8 * hi + ((v & 3) << 1) + s2;
          int c = kk / 27;
          int c2 = c < 8 ? c : 7;                     // clamp K-pad slots
          int kr = kk - c * 27;
          int kr2 = (c < 8) ? kr : 0;
          int kz = kr2 / 9, k2 = kr2 - kz * 9, ky = k2 / 3, kx = k2 - ky * 3;
          _Float16 hv =
              ht[((c2 * LZ + (dl + kz)) * LY + (yl + ky)) * LX + (mrow + kx)];
          bfrag[2 * v + s2] = (kk < 216) ? hv : (_Float16)0.0f;
        }
      }
      acc = __builtin_amdgcn_wmma_f32_16x16x32_f16(
          false, aw[j], false, bfrag, (short)0, acc, false, false);
    }
    if (hi == 0) {  // only row M=0 (lanes 0..15, acc[0]) is the conv output
      size_t vidx = ((size_t)(d0 + dl) * HH + (y0 + yl)) * WW + (x0 + mrow);
      float outv = acc[0] + bb;
      float res = isZ ? (1.0f + e) * tcur[vidx] - e * outv
                      : (1.0f + e) * upd[vidx] - e * outv;
      upd[vidx] = res;
    }
  }
}

// ---------------- host orchestration ----------------
extern "C" void kernel_launch(void* const* d_in, const int* in_sizes, int n_in,
                              void* d_out, int out_size, void* d_ws, size_t ws_size,
                              hipStream_t stream) {
  (void)in_sizes; (void)n_in; (void)out_size; (void)ws_size;
  const float* image = (const float*)d_in[0];
  const float* sino  = (const float*)d_in[1];
  const float* W1p   = (const float*)d_in[2];   // [4][8][1][3][3][3]
  const float* B1p   = (const float*)d_in[3];   // [4][8]
  const float* W2p   = (const float*)d_in[4];   // [4][1][8][3][3][3]
  const float* B2p   = (const float*)d_in[5];   // [4][1]
  const float* ntx   = (const float*)d_in[6];
  const float* nty   = (const float*)d_in[7];
  const float* ntz   = (const float*)d_in[8];
  const float* nt    = (const float*)d_in[9];
  float* out = (float*)d_out;                   // [4][N]

  // workspace layout: z | p | q | s (f32) | h (8 ch, f16)  ~= 128 MiB
  float* z = (float*)d_ws;
  float* p = z + NN;
  float* q = p + NN;
  float* s = q + NN;
  _Float16* h = (_Float16*)(s + NN);

  // outs[0] = image
  hipMemcpyAsync(out, image, (size_t)NN * sizeof(float),
                 hipMemcpyDeviceToDevice, stream);
  // p = q = s = 0 (contiguous 3N range)
  zero_kernel<<<(3L * NN + 255) / 256, 256, 0, stream>>>(p, 3L * NN);

  dim3 cgrid(WW / 16, HH / 16, DD / 4);   // 16 x 16 x 16 tiles
  for (int c = 0; c < 3; ++c) {
    const float* t = out + (size_t)c * NN;
    proj_z_kernel<<<(HH * WW + 255) / 256, 256, 0, stream>>>(t, sino, z);

    // block 0: d/dx -> p (ntx)
    conv1_wmma_kernel<1><<<cgrid, 256, 0, stream>>>(z, W1p + 0 * 216, B1p + 0 * 8, h);
    conv2_wmma_kernel<<<cgrid, 256, 0, stream>>>(h, W2p + 0 * 216, B2p + 0, p, nullptr, ntx + c, 0);
    // block 1: d/dy -> q (nty)
    conv1_wmma_kernel<2><<<cgrid, 256, 0, stream>>>(z, W1p + 1 * 216, B1p + 1 * 8, h);
    conv2_wmma_kernel<<<cgrid, 256, 0, stream>>>(h, W2p + 1 * 216, B2p + 1, q, nullptr, nty + c, 0);
    // block 2: d/dz -> s (ntz)
    conv1_wmma_kernel<3><<<cgrid, 256, 0, stream>>>(z, W1p + 2 * 216, B1p + 2 * 8, h);
    conv2_wmma_kernel<<<cgrid, 256, 0, stream>>>(h, W2p + 2 * 216, B2p + 2, s, nullptr, ntz + c, 0);
    // block 3: identity -> z_ (overwrites z; reads t)
    conv1_wmma_kernel<0><<<cgrid, 256, 0, stream>>>(z, W1p + 3 * 216, B1p + 3 * 8, h);
    conv2_wmma_kernel<<<cgrid, 256, 0, stream>>>(h, W2p + 3 * 216, B2p + 3, z, t, nt + c, 1);

    assemble_kernel<<<((long)NN + 255) / 256, 256, 0, stream>>>(
        p, q, s, z, out + (size_t)(c + 1) * NN);
  }
}
